// SwinMSA_27212912787508
// MI455X (gfx1250) — matrix operand order repaired and checked by reading, compiled, not run on hardware
//
#include <hip/hip_runtime.h>
#include <hip/hip_bf16.h>

typedef __bf16 v16bf __attribute__((ext_vector_type(16)));
typedef __bf16 v8bf  __attribute__((ext_vector_type(8)));
typedef float  v8f   __attribute__((ext_vector_type(8)));

#define WMMA_BF16(a, b, c) \
    __builtin_amdgcn_wmma_f32_16x16x32_bf16(false, (a), false, (b), (short)0, (c), false, false)

// Combine the two contiguous 8x bf16 runs of an A fragment (elements 0-7 and
// 8-15, K = 16*half + i) into one v16bf operand.
__device__ __forceinline__ v16bf cat16(v8bf lo, v8bf hi) {
    return __builtin_shufflevector(lo, hi, 0, 1, 2, 3, 4, 5, 6, 7,
                                           8, 9, 10, 11, 12, 13, 14, 15);
}

// ---------------------------------------------------------------------------
// Kernel 0: pack W_qkv (512x1536 f32) and W_proj (512x512 f32) into bf16
// WMMA-B fragments. Fragment layout (per 16x16x32 bf16 WMMA B = 32x16):
//   lane 0-15 : column N = lane,      element i -> K = i        (K 0..15)
//   lane 16-31: column N = lane - 16, element i -> K = 16 + i   (K 16..31)
// Packed index: (((ntile*16 + kstep)*32 + lane)*16 + i), 32B contiguous/lane.
// ---------------------------------------------------------------------------
__global__ void pack_weights_kernel(const float* __restrict__ Wq,
                                    const float* __restrict__ Wp,
                                    __bf16* __restrict__ wsQ,
                                    __bf16* __restrict__ wsP) {
    const int NQ = 96 * 16 * 32 * 16;   // 786432 elements for W_qkv
    const int NP = 32 * 16 * 32 * 16;   // 262144 for W_proj
    int idx = blockIdx.x * blockDim.x + threadIdx.x;
    if (idx < NQ) {
        int i    = idx & 15;
        int lane = (idx >> 4) & 31;
        int k    = (idx >> 9) & 15;
        int nt   = idx >> 13;
        int K    = k * 32 + (lane >> 4) * 16 + i;
        int col  = nt * 16 + (lane & 15);
        wsQ[idx] = (__bf16)Wq[K * 1536 + col];
    } else if (idx < NQ + NP) {
        int j    = idx - NQ;
        int i    = j & 15;
        int lane = (j >> 4) & 31;
        int k    = (j >> 9) & 15;
        int nt   = j >> 13;
        int K    = k * 32 + (lane >> 4) * 16 + i;
        int col  = nt * 16 + (lane & 15);
        wsP[j]   = (__bf16)Wp[K * 512 + col];
    }
}

// ---------------------------------------------------------------------------
// Main fused kernel: one block per batch element b. 256 threads = 8 waves.
// Wave w processes heads {w, w+8} end-to-end (no cross-wave deps), then all
// waves cooperate on the projection GEMM after one barrier.
// Per-wave LDS: Wqk = 64x64 bf16 (q cols 0..31 | k cols 32..63), later reused
// for the attention probabilities; vT = 32x64 bf16 = V transposed [e][krow].
// ---------------------------------------------------------------------------
__global__ __launch_bounds__(256) void msa_fused_kernel(
    const float* __restrict__ x,      // [B, 64, 512]
    const float* __restrict__ mask,   // [64, 1, 64]
    const float* __restrict__ bqkv,   // [1536]
    const float* __restrict__ bproj,  // [512]
    const __bf16* __restrict__ wsQ,   // packed W_qkv B-fragments
    const __bf16* __restrict__ wsP,   // packed W_proj B-fragments
    float* __restrict__ out)          // [B, 64, 512]
{
    __shared__ __bf16 qk_s[8 * 64 * 64];  // per-wave q|k (8 KB each)
    __shared__ __bf16 vt_s[8 * 32 * 64];  // per-wave V^T  (4 KB each)
    __shared__ __bf16 ao[64 * 512];       // attention output, bf16, row-major

    const int b    = blockIdx.x;
    const int tid  = threadIdx.x;
    const int wave = tid >> 5;
    const int lane = tid & 31;
    const int r16  = lane & 15;     // tile column / row-in-half
    const int half = lane >> 4;     // 16-lane half of the wave
    const float scale = 0.17677669529663687f;  // 1/sqrt(32)

    __bf16* Wqk = qk_s + wave * 64 * 64;
    __bf16* vT  = vt_s + wave * 32 * 64;

    for (int hp = 0; hp < 2; ++hp) {
        const int head = hp * 8 + wave;

        // ---------- QKV GEMM for this head: M=64, N=96 (6 tiles), K=512 ----
        for (int m = 0; m < 4; ++m) {
            v8f acc[6];
#pragma unroll
            for (int n = 0; n < 6; ++n) {
                float bv = bqkv[head * 96 + n * 16 + r16];
#pragma unroll
                for (int r = 0; r < 8; ++r) acc[n][r] = bv;
            }
            const float* xrow = x + ((size_t)b * 64 + m * 16 + r16) * 512;
            for (int k = 0; k < 16; ++k) {
                // A fragment (16x32 bf16) streamed from global, fp32->bf16.
                const float4* p0 = (const float4*)(xrow + k * 32 + half * 16);
                const float4* p1 = (const float4*)(xrow + k * 32 + half * 16 + 8);
                float4 f0 = p0[0], f1 = p0[1], f2 = p1[0], f3 = p1[1];
                v16bf a;
                a[0]  = (__bf16)f0.x; a[1]  = (__bf16)f0.y;
                a[2]  = (__bf16)f0.z; a[3]  = (__bf16)f0.w;
                a[4]  = (__bf16)f1.x; a[5]  = (__bf16)f1.y;
                a[6]  = (__bf16)f1.z; a[7]  = (__bf16)f1.w;
                a[8]  = (__bf16)f2.x; a[9]  = (__bf16)f2.y;
                a[10] = (__bf16)f2.z; a[11] = (__bf16)f2.w;
                a[12] = (__bf16)f3.x; a[13] = (__bf16)f3.y;
                a[14] = (__bf16)f3.z; a[15] = (__bf16)f3.w;
#pragma unroll
                for (int n = 0; n < 6; ++n) {
                    v16bf bf = *(const v16bf*)(wsQ +
                        ((((size_t)(head * 6 + n) * 16 + k) * 32 + lane) << 4));
                    acc[n] = WMMA_BF16(a, bf, acc[n]);
                }
            }
            // q (n=0,1) and k (n=2,3) -> row-major Wqk
#pragma unroll
            for (int n = 0; n < 4; ++n)
#pragma unroll
                for (int r = 0; r < 8; ++r)
                    Wqk[(m * 16 + r + 8 * half) * 64 + n * 16 + r16] =
                        (__bf16)acc[n][r];
            // v (n=4,5) -> transposed vT[e][krow]; r=0..7 contiguous -> 16B store
#pragma unroll
            for (int n = 4; n < 6; ++n) {
                v8bf pv;
#pragma unroll
                for (int r = 0; r < 8; ++r) pv[r] = (__bf16)acc[n][r];
                *(v8bf*)(vT + ((n - 4) * 16 + r16) * 64 + m * 16 + 8 * half) = pv;
            }
        }

        // ---------- energy = q @ k^T : M=64, N=64, K=32 ---------------------
        float e[4][4][8];
#pragma unroll
        for (int m = 0; m < 4; ++m) {
            v8bf alo = *(const v8bf*)(Wqk + (m * 16 + r16) * 64 + half * 16);
            v8bf ahi = *(const v8bf*)(Wqk + (m * 16 + r16) * 64 + half * 16 + 8);
            v16bf a = cat16(alo, ahi);
#pragma unroll
            for (int n = 0; n < 4; ++n) {
                // B = k^T: lane column = k-row (n*16+r16), element i -> e dim
                v16bf bf = *(const v16bf*)(Wqk + (n * 16 + r16) * 64 + 32 + half * 16);
                v8f c = {};
                c = WMMA_BF16(a, bf, c);
#pragma unroll
                for (int r = 0; r < 8; ++r)
                    e[m][n][r] = c[r] * scale +
                        mask[(m * 16 + r + 8 * half) * 64 + n * 16 + r16];
            }
        }

        // ---------- row softmax over 64 keys (rows live in 16-lane halves) -
#pragma unroll
        for (int m = 0; m < 4; ++m)
#pragma unroll
            for (int r = 0; r < 8; ++r) {
                float t = fmaxf(fmaxf(e[m][0][r], e[m][1][r]),
                                fmaxf(e[m][2][r], e[m][3][r]));
#pragma unroll
                for (int off = 8; off >= 1; off >>= 1)
                    t = fmaxf(t, __shfl_xor(t, off, 16));
                float s = 0.0f;
#pragma unroll
                for (int n = 0; n < 4; ++n) {
                    e[m][n][r] = __expf(e[m][n][r] - t);
                    s += e[m][n][r];
                }
#pragma unroll
                for (int off = 8; off >= 1; off >>= 1)
                    s += __shfl_xor(s, off, 16);
                float rinv = 1.0f / s;
                // probs overwrite the (now dead) q|k region as bf16, 64x64
#pragma unroll
                for (int n = 0; n < 4; ++n)
                    Wqk[(m * 16 + r + 8 * half) * 64 + n * 16 + r16] =
                        (__bf16)(e[m][n][r] * rinv);
            }

        // ---------- out_head = attn @ v : M=64, N=32, K=64 ------------------
#pragma unroll
        for (int m = 0; m < 4; ++m)
#pragma unroll
            for (int n2 = 0; n2 < 2; ++n2) {
                v8f c = {};
#pragma unroll
                for (int k2 = 0; k2 < 2; ++k2) {
                    v8bf alo = *(const v8bf*)(Wqk + (m * 16 + r16) * 64 +
                                              k2 * 32 + half * 16);
                    v8bf ahi = *(const v8bf*)(Wqk + (m * 16 + r16) * 64 +
                                              k2 * 32 + half * 16 + 8);
                    // B[K=krow][N=e] from vT[e][krow]: contiguous 32B per lane
                    v16bf bf = *(const v16bf*)(vT + (n2 * 16 + r16) * 64 +
                                               k2 * 32 + half * 16);
                    c = WMMA_BF16(cat16(alo, ahi), bf, c);
                }
#pragma unroll
                for (int r = 0; r < 8; ++r)
                    ao[(m * 16 + r + 8 * half) * 512 +
                       head * 32 + n2 * 16 + r16] = (__bf16)c[r];
            }
    }

    __syncthreads();

    // ---------- projection: out = ao @ W_proj + b_proj : 64 x 512, K=512 ----
    // wave w owns N-tiles [4w, 4w+4)
    for (int m = 0; m < 4; ++m) {
        v8f acc[4];
#pragma unroll
        for (int nn = 0; nn < 4; ++nn) {
            float bv = bproj[(wave * 4 + nn) * 16 + r16];
#pragma unroll
            for (int r = 0; r < 8; ++r) acc[nn][r] = bv;
        }
        for (int k = 0; k < 16; ++k) {
            v8bf alo = *(const v8bf*)(ao + (m * 16 + r16) * 512 +
                                      k * 32 + half * 16);
            v8bf ahi = *(const v8bf*)(ao + (m * 16 + r16) * 512 +
                                      k * 32 + half * 16 + 8);
            v16bf a = cat16(alo, ahi);
#pragma unroll
            for (int nn = 0; nn < 4; ++nn) {
                int n = wave * 4 + nn;
                v16bf bf = *(const v16bf*)(wsP +
                    ((((size_t)n * 16 + k) * 32 + lane) << 4));
                acc[nn] = WMMA_BF16(a, bf, acc[nn]);
            }
        }
#pragma unroll
        for (int nn = 0; nn < 4; ++nn)
#pragma unroll
            for (int r = 0; r < 8; ++r)
                out[((size_t)b * 64 + m * 16 + r + 8 * half) * 512 +
                    (wave * 4 + nn) * 16 + r16] = acc[nn][r];
    }
}

// ---------------------------------------------------------------------------
extern "C" void kernel_launch(void* const* d_in, const int* in_sizes, int n_in,
                              void* d_out, int out_size, void* d_ws, size_t ws_size,
                              hipStream_t stream) {
    const float* x    = (const float*)d_in[0];
    const float* mask = (const float*)d_in[1];
    const float* Wq   = (const float*)d_in[2];
    const float* bq   = (const float*)d_in[3];
    const float* Wp   = (const float*)d_in[4];
    const float* bp   = (const float*)d_in[5];
    float* out        = (float*)d_out;

    __bf16* wsQ = (__bf16*)d_ws;            // 786432 bf16 = 1.5 MB
    __bf16* wsP = wsQ + 96 * 16 * 32 * 16;  // 262144 bf16 = 0.5 MB

    const int B = in_sizes[0] / (64 * 512);  // 2048

    const int packN = (96 * 16 * 32 * 16) + (32 * 16 * 32 * 16);
    pack_weights_kernel<<<(packN + 255) / 256, 256, 0, stream>>>(Wq, Wp, wsQ, wsP);
    msa_fused_kernel<<<B, 256, 0, stream>>>(x, mask, bq, bp, wsQ, wsP, out);
}
